// FacePartGAT_79963701117661
// MI455X (gfx1250) — matrix operand order repaired and checked by reading, not compiled, not run on hardware
//
#include <hip/hip_runtime.h>
#include <hip/hip_bf16.h>
#include <math.h>

typedef __attribute__((ext_vector_type(16))) _Float16 v16h;
typedef __attribute__((ext_vector_type(8)))  _Float16 v8h;
typedef __attribute__((ext_vector_type(8)))  float    v8f;

#define NNODE 4096
#define HIDD  768

#define WMMA_F16(a, b, c) \
  __builtin_amdgcn_wmma_f32_16x16x32_f16(false, (a), false, (b), (short)0, (c), false, false)

// ---------------------------------------------------------------------------
// WMMA operand loaders (16-bit operands, wave32; contiguous k order, same for A and B)
// A 16x32: lane L -> row M=L%16; elems 0-15 <- K = 2*off+e (contiguous),
//          off = (L<16 ? 0 : 8).
// B 32x16: lane L -> col N=L%16; elems 0-15 <- K = (L<16?0:16)+e (contiguous).
// ---------------------------------------------------------------------------
__device__ __forceinline__ v16h load_a16(const _Float16* rowptr, int off) {
  v8h lo = *(const v8h*)(rowptr + 2 * off);
  v8h hi = *(const v8h*)(rowptr + 2 * off + 8);
  v16h a;
#pragma unroll
  for (int i = 0; i < 8; ++i) { a[i] = lo[i]; a[i + 8] = hi[i]; }
  return a;
}

__device__ __forceinline__ v16h load_b16(const _Float16* p) {
  v8h lo = *(const v8h*)(p);
  v8h hi = *(const v8h*)(p + 8);
  v16h b;
#pragma unroll
  for (int i = 0; i < 8; ++i) { b[i] = lo[i]; b[i + 8] = hi[i]; }
  return b;
}

// ---------------------------------------------------------------------------
// f32 -> f16 convert / transpose-convert (weights -> "N x K" B-operand layout)
// ---------------------------------------------------------------------------
__global__ void k_f32_to_f16(const float* __restrict__ src, _Float16* __restrict__ dst, int n) {
  int i = blockIdx.x * 256 + threadIdx.x;
  if (i < n) dst[i] = (_Float16)src[i];
}

__global__ void k_transpose_f16(const float* __restrict__ src, _Float16* __restrict__ dst,
                                int R, int C) {  // src RxC row-major -> dst CxR f16
  int i = blockIdx.x * 256 + threadIdx.x;
  if (i < R * C) {
    int r = i / C, c = i % C;
    dst[(size_t)c * R + r] = (_Float16)src[i];
  }
}

// ---------------------------------------------------------------------------
// Generic WMMA GEMM: C = A(f16 MxK) @ B (BT f16 stored NxK) [+ bias]
// block = 256 (8 waves); wave w -> 16x16 tile at (blockIdx.x*16, blockIdx.y*128+w*16)
// MODE 0: write f32 row-major + f16 transposed (N x NNODE) for next-stage B operand
// MODE 1: qkv epilogue: q,k f16 row-major (4096x128), v f16 transposed (128x4096)
// ---------------------------------------------------------------------------
template <int MODE>
__global__ void k_gemm(const _Float16* __restrict__ A, int lda,
                       const _Float16* __restrict__ BT, int ldb, const float* __restrict__ bias,
                       int K, float* __restrict__ Cf32, int ldc, _Float16* __restrict__ CT,
                       _Float16* __restrict__ Qf, _Float16* __restrict__ Kf,
                       _Float16* __restrict__ VT) {
  const int lane = threadIdx.x & 31;
  const int wave = threadIdx.x >> 5;
  const int m0 = blockIdx.x * 16;
  const int n0 = blockIdx.y * 128 + wave * 16;
  const int hs = (lane < 16) ? 0 : 1;
  const int lmod = lane & 15;
  const _Float16* arow = A + (size_t)(m0 + lmod) * lda;
  const _Float16* brow = BT + (size_t)(n0 + lmod) * ldb;
  v8f c = {};
  for (int k0 = 0; k0 < K; k0 += 32) {
    __builtin_prefetch(arow + k0 + 64, 0, 1);
    v16h a = load_a16(arow + k0, hs * 8);
    v16h b = load_b16(brow + k0 + hs * 16);
    c = WMMA_F16(a, b, c);
  }
  const int ncol = n0 + lmod;
  const float bv = bias ? bias[ncol] : 0.f;
#pragma unroll
  for (int r = 0; r < 8; ++r) {
    const int m = m0 + r + hs * 8;
    const float v = c[r] + bv;
    if (MODE == 0) {
      Cf32[(size_t)m * ldc + ncol] = v;
      CT[(size_t)ncol * NNODE + m] = (_Float16)v;
    } else {
      if (ncol < 128)       Qf[(size_t)m * 128 + ncol] = (_Float16)v;
      else if (ncol < 256)  Kf[(size_t)m * 128 + (ncol - 128)] = (_Float16)v;
      else                  VT[(size_t)(ncol - 256) * NNODE + m] = (_Float16)v;
    }
  }
}

// ---------------------------------------------------------------------------
// GAT attention scalars: es[h][n] = h_n . a_src[h], ed likewise (ch = 128)
// block = heads*32
// ---------------------------------------------------------------------------
__global__ void k_esed(const float* __restrict__ h, const float* __restrict__ asrc,
                       const float* __restrict__ adst, float* __restrict__ es,
                       float* __restrict__ ed, int heads) {
  const int t = threadIdx.x, head = t >> 5, lane = t & 31, n = blockIdx.x;
  const float* base = h + (size_t)n * heads * 128 + head * 128;
  float s = 0.f, d = 0.f;
  for (int c = lane; c < 128; c += 32) {
    float v = base[c];
    s += v * asrc[head * 128 + c];
    d += v * adst[head * 128 + c];
  }
#pragma unroll
  for (int m = 16; m; m >>= 1) { s += __shfl_xor(s, m, 32); d += __shfl_xor(d, m, 32); }
  if (lane == 0) { es[head * NNODE + n] = s; ed[head * NNODE + n] = d; }
}

__global__ void k_colmax(const float* __restrict__ es, float* __restrict__ m, int n) {
  __shared__ float red[256];
  const int head = blockIdx.x, t = threadIdx.x;
  float lm = -3.4e38f;
  for (int i = t; i < n; i += 256) lm = fmaxf(lm, es[head * n + i]);
  red[t] = lm; __syncthreads();
  for (int s = 128; s; s >>= 1) { if (t < s) red[t] = fmaxf(red[t], red[t + s]); __syncthreads(); }
  if (t == 0) m[head] = red[0];
}

// ---------------------------------------------------------------------------
// GAT aggregation (rank-1 softmax, closed-form row max):
//   out_j = (sum_i exp(lrelu(ed_j+es_i)-M_j) * h_i) / denom_j ; +bias, ELU, f16
// grid(NNODE/16, heads), block 256 (8 waves = 8 col tiles of the 128-ch head)
// ---------------------------------------------------------------------------
__global__ void k_gat_agg(const float* __restrict__ es, const float* __restrict__ ed,
                          const float* __restrict__ mvec, const _Float16* __restrict__ hT,
                          const float* __restrict__ bias, _Float16* __restrict__ xout,
                          int ldout) {
  __shared__ float edL[16];
  __shared__ float rowM[16];
  __shared__ __align__(16) _Float16 wL[16 * 32];
  __shared__ float dnL[16];
  const int t = threadIdx.x, lane = t & 31, wave = t >> 5;
  const int head = blockIdx.y, m0 = blockIdx.x * 16;
  const int colbase = head * 128;
  const int hs = (lane < 16) ? 0 : 1;
  const int lmod = lane & 15;
  if (t < 16) {
    float e = ed[head * NNODE + m0 + t];
    edL[t] = e;
    float mm = e + mvec[head];
    rowM[t] = (mm > 0.f) ? mm : 0.2f * mm;  // lrelu(ed_j + max_i es_i)
  }
  __syncthreads();
  v8f c = {};
  float dpart = 0.f;
  const _Float16* bbase = hT + (size_t)(colbase + wave * 16 + lmod) * NNODE;
  const float* esrow = es + (size_t)head * NNODE;
  for (int k0 = 0; k0 < NNODE; k0 += 32) {
#pragma unroll
    for (int i = 0; i < 2; ++i) {  // 512 weights / 256 threads
      const int idx = t + i * 256;
      const int row = idx >> 5, col = idx & 31;
      float e = edL[row] + esrow[k0 + col];
      e = (e > 0.f) ? e : 0.2f * e;
      wL[row * 32 + col] = (_Float16)__expf(e - rowM[row]);
    }
    __syncthreads();
    if (t < 16) {
      float s = 0.f;
#pragma unroll
      for (int col = 0; col < 32; ++col) s += (float)wL[t * 32 + col];
      dpart += s;
    }
    v16h a = load_a16(&wL[lmod * 32], hs * 8);
    v16h b = load_b16(bbase + k0 + hs * 16);
    c = WMMA_F16(a, b, c);
    __syncthreads();
  }
  if (t < 16) dnL[t] = dpart;
  __syncthreads();
  const int ncol = colbase + wave * 16 + lmod;
  const float bv = bias[ncol];
#pragma unroll
  for (int r = 0; r < 8; ++r) {
    const int row = r + hs * 8;
    float v = c[r] / dnL[row] + bv;
    v = (v > 0.f) ? v : (__expf(v) - 1.f);  // ELU
    xout[(size_t)(m0 + row) * ldout + ncol] = (_Float16)v;
  }
}

// ---------------------------------------------------------------------------
// MHA pass 1: rowmax[h][j] = max_i (q_j . k_i) * scale ; grid(N/16, H), 1 wave
// ---------------------------------------------------------------------------
__global__ void k_mha_rowmax(const _Float16* __restrict__ q, const _Float16* __restrict__ kk,
                             float* __restrict__ rowmax) {
  const int lane = threadIdx.x & 31;
  const int head = blockIdx.y, m0 = blockIdx.x * 16;
  const int hs = (lane < 16) ? 0 : 1;
  const int lmod = lane & 15;
  const v16h aq = load_a16(q + (size_t)(m0 + lmod) * 128 + head * 32, hs * 8);
  float mreg[8];
#pragma unroll
  for (int r = 0; r < 8; ++r) mreg[r] = -3.4e38f;
  for (int s0 = 0; s0 < NNODE; s0 += 16) {
    v16h b = load_b16(kk + (size_t)(s0 + lmod) * 128 + head * 32 + hs * 16);
    v8f c = {};
    c = WMMA_F16(aq, b, c);
#pragma unroll
    for (int r = 0; r < 8; ++r) mreg[r] = fmaxf(mreg[r], c[r]);
  }
#pragma unroll
  for (int r = 0; r < 8; ++r) {
#pragma unroll
    for (int m = 8; m; m >>= 1) mreg[r] = fmaxf(mreg[r], __shfl_xor(mreg[r], m, 32));
  }
  const float scale = 0.17677669529663687f;  // 1/sqrt(32)
  if (lane == 0) {
#pragma unroll
    for (int r = 0; r < 8; ++r) rowmax[head * NNODE + m0 + r] = mreg[r] * scale;
  }
  if (lane == 16) {
#pragma unroll
    for (int r = 0; r < 8; ++r) rowmax[head * NNODE + m0 + 8 + r] = mreg[r] * scale;
  }
}

// ---------------------------------------------------------------------------
// MHA pass 2: flash-style softmax(qk)*v with precomputed row max; 1 wave/block
// ---------------------------------------------------------------------------
__global__ void k_mha_flash(const _Float16* __restrict__ q, const _Float16* __restrict__ kk,
                            const _Float16* __restrict__ vT, const float* __restrict__ rowmax,
                            float* __restrict__ aout) {
  __shared__ float sL[16 * 32];
  __shared__ __align__(16) _Float16 wL[16 * 32];
  __shared__ float rm[16];
  __shared__ float dn[16];
  const int lane = threadIdx.x & 31;
  const int head = blockIdx.y, m0 = blockIdx.x * 16;
  const int hs = (lane < 16) ? 0 : 1;
  const int lmod = lane & 15;
  const float scale = 0.17677669529663687f;
  if (lane < 16) rm[lane] = rowmax[head * NNODE + m0 + lane];
  __syncthreads();
  const v16h aq = load_a16(q + (size_t)(m0 + lmod) * 128 + head * 32, hs * 8);
  v8f o0 = {}, o1 = {};
  float dpart = 0.f;
  for (int s0 = 0; s0 < NNODE; s0 += 32) {
    v8f ca = {}, cb = {};
    v16h b0 = load_b16(kk + (size_t)(s0 + lmod) * 128 + head * 32 + hs * 16);
    ca = WMMA_F16(aq, b0, ca);
    v16h b1 = load_b16(kk + (size_t)(s0 + 16 + lmod) * 128 + head * 32 + hs * 16);
    cb = WMMA_F16(aq, b1, cb);
#pragma unroll
    for (int r = 0; r < 8; ++r) {
      const int row = r + hs * 8;
      sL[row * 32 + lmod] = ca[r] * scale;
      sL[row * 32 + 16 + lmod] = cb[r] * scale;
    }
    __syncthreads();
    {
      const int row = lmod, cb0 = hs * 16;
#pragma unroll
      for (int j = 0; j < 16; ++j) {
        float w = __expf(sL[row * 32 + cb0 + j] - rm[row]);
        wL[row * 32 + cb0 + j] = (_Float16)w;
        dpart += w;
      }
    }
    __syncthreads();
    v16h a = load_a16(&wL[lmod * 32], hs * 8);
    v16h bv0 = load_b16(vT + (size_t)(head * 32 + lmod) * NNODE + s0 + hs * 16);
    o0 = WMMA_F16(a, bv0, o0);
    v16h bv1 = load_b16(vT + (size_t)(head * 32 + 16 + lmod) * NNODE + s0 + hs * 16);
    o1 = WMMA_F16(a, bv1, o1);
    __syncthreads();
  }
  float dfull = dpart + __shfl_xor(dpart, 16, 32);
  if (lane < 16) dn[lane] = dfull;
  __syncthreads();
#pragma unroll
  for (int r = 0; r < 8; ++r) {
    const int row = r + hs * 8;
    const float inv = 1.f / dn[row];
    aout[(size_t)(m0 + row) * 128 + head * 32 + lmod] = o0[r] * inv;
    aout[(size_t)(m0 + row) * 128 + head * 32 + 16 + lmod] = o1[r] * inv;
  }
}

// ---------------------------------------------------------------------------
// Column mean of attn output (mean over tokens commutes with out_proj)
// ---------------------------------------------------------------------------
__global__ void k_colmean(const float* __restrict__ a, float* __restrict__ mbar) {
  __shared__ float red[256];
  const int c = blockIdx.x, t = threadIdx.x;
  float s = 0.f;
  for (int i = t; i < NNODE; i += 256) s += a[(size_t)i * 128 + c];
  red[t] = s; __syncthreads();
  for (int st = 128; st; st >>= 1) { if (t < st) red[t] += red[t + st]; __syncthreads(); }
  if (t == 0) mbar[c] = red[0] * (1.f / NNODE);
}

// ---------------------------------------------------------------------------
// Tail: pooled = mbar@out_w+out_b ; p2 = pooled@fc_w+fc_b ; out = p2@final_w+b
// ---------------------------------------------------------------------------
__global__ void k_head(const float* __restrict__ mbar, const float* __restrict__ ow,
                       const float* __restrict__ ob, const float* __restrict__ fw,
                       const float* __restrict__ fb, const float* __restrict__ Fw,
                       const float* __restrict__ Fb, float* __restrict__ out) {
  __shared__ float m0s[128], p1[128], p2[128];
  const int t = threadIdx.x;
  if (t < 128) m0s[t] = mbar[t];
  __syncthreads();
  if (t < 128) {
    float acc = ob[t];
    for (int k = 0; k < 128; ++k) acc += m0s[k] * ow[k * 128 + t];
    p1[t] = acc;
  }
  __syncthreads();
  if (t < 128) {
    float acc = fb[t];
    for (int k = 0; k < 128; ++k) acc += p1[k] * fw[k * 128 + t];
    p2[t] = acc;
  }
  __syncthreads();
  for (int j = t; j < HIDD; j += 256) {
    float acc = Fb[j];
    for (int k = 0; k < 128; ++k) acc += p2[k] * Fw[k * HIDD + j];
    out[j] = acc;
  }
}

// ---------------------------------------------------------------------------
extern "C" void kernel_launch(void* const* d_in, const int* in_sizes, int n_in,
                              void* d_out, int out_size, void* d_ws, size_t ws_size,
                              hipStream_t stream) {
  const float* node   = (const float*)d_in[0];
  const float* W1     = (const float*)d_in[1];
  const float* asrc1  = (const float*)d_in[2];
  const float* adst1  = (const float*)d_in[3];
  const float* b1     = (const float*)d_in[4];
  const float* W2     = (const float*)d_in[5];
  const float* asrc2  = (const float*)d_in[6];
  const float* adst2  = (const float*)d_in[7];
  const float* b2     = (const float*)d_in[8];
  const float* inW    = (const float*)d_in[9];
  const float* inB    = (const float*)d_in[10];
  const float* outW   = (const float*)d_in[11];
  const float* outB   = (const float*)d_in[12];
  const float* fcW    = (const float*)d_in[13];
  const float* fcB    = (const float*)d_in[14];
  const float* finW   = (const float*)d_in[15];
  const float* finB   = (const float*)d_in[16];
  float* outp = (float*)d_out;

  char* ws = (char*)d_ws;
  size_t off = 0;
  auto alloc = [&](size_t bytes) -> char* {
    off = (off + 255) & ~(size_t)255;
    char* p = ws + off;
    off += bytes;
    return p;
  };
  _Float16* xf16 = (_Float16*)alloc((size_t)NNODE * HIDD * 2);
  _Float16* W1T  = (_Float16*)alloc((size_t)512 * 768 * 2);
  _Float16* W2T  = (_Float16*)alloc((size_t)128 * 512 * 2);
  _Float16* inT  = (_Float16*)alloc((size_t)384 * 128 * 2);
  float*    h1   = (float*)   alloc((size_t)NNODE * 512 * 4);
  _Float16* h1T  = (_Float16*)alloc((size_t)512 * NNODE * 2);
  float*    es1  = (float*)   alloc((size_t)4 * NNODE * 4);
  float*    ed1  = (float*)   alloc((size_t)4 * NNODE * 4);
  float*    m1   = (float*)   alloc(4 * 4);
  _Float16* x1   = (_Float16*)alloc((size_t)NNODE * 512 * 2);
  float*    h2   = (float*)   alloc((size_t)NNODE * 128 * 4);
  _Float16* h2T  = (_Float16*)alloc((size_t)128 * NNODE * 2);
  float*    es2  = (float*)   alloc((size_t)NNODE * 4);
  float*    ed2  = (float*)   alloc((size_t)NNODE * 4);
  float*    m2   = (float*)   alloc(4);
  _Float16* x2   = (_Float16*)alloc((size_t)NNODE * 128 * 2);
  _Float16* qf   = (_Float16*)alloc((size_t)NNODE * 128 * 2);
  _Float16* kf   = (_Float16*)alloc((size_t)NNODE * 128 * 2);
  _Float16* vT   = (_Float16*)alloc((size_t)128 * NNODE * 2);
  float*    rmax = (float*)   alloc((size_t)4 * NNODE * 4);
  float*    aout = (float*)   alloc((size_t)NNODE * 128 * 4);
  float*    mbar = (float*)   alloc(128 * 4);

  // --- prep: convert inputs, transpose weights to (N x K) f16 B-operand form
  k_f32_to_f16<<<(NNODE * HIDD + 255) / 256, 256, 0, stream>>>(node, xf16, NNODE * HIDD);
  k_transpose_f16<<<(768 * 512 + 255) / 256, 256, 0, stream>>>(W1, W1T, 768, 512);
  k_transpose_f16<<<(512 * 128 + 255) / 256, 256, 0, stream>>>(W2, W2T, 512, 128);
  k_transpose_f16<<<(128 * 384 + 255) / 256, 256, 0, stream>>>(inW, inT, 128, 384);

  // --- GAT layer 1
  k_gemm<0><<<dim3(NNODE / 16, 4), 256, 0, stream>>>(xf16, HIDD, W1T, HIDD, nullptr, HIDD,
                                                     h1, 512, h1T, nullptr, nullptr, nullptr);
  k_esed<<<NNODE, 4 * 32, 0, stream>>>(h1, asrc1, adst1, es1, ed1, 4);
  k_colmax<<<4, 256, 0, stream>>>(es1, m1, NNODE);
  k_gat_agg<<<dim3(NNODE / 16, 4), 256, 0, stream>>>(es1, ed1, m1, h1T, b1, x1, 512);

  // --- GAT layer 2
  k_gemm<0><<<dim3(NNODE / 16, 1), 256, 0, stream>>>(x1, 512, W2T, 512, nullptr, 512,
                                                     h2, 128, h2T, nullptr, nullptr, nullptr);
  k_esed<<<NNODE, 32, 0, stream>>>(h2, asrc2, adst2, es2, ed2, 1);
  k_colmax<<<1, 256, 0, stream>>>(es2, m2, NNODE);
  k_gat_agg<<<dim3(NNODE / 16, 1), 256, 0, stream>>>(es2, ed2, m2, h2T, b2, x2, 128);

  // --- MHA pooling
  k_gemm<1><<<dim3(NNODE / 16, 3), 256, 0, stream>>>(x2, 128, inT, 128, inB, 128,
                                                     nullptr, 0, nullptr, qf, kf, vT);
  k_mha_rowmax<<<dim3(NNODE / 16, 4), 32, 0, stream>>>(qf, kf, rmax);
  k_mha_flash<<<dim3(NNODE / 16, 4), 32, 0, stream>>>(qf, kf, vT, rmax, aout);
  k_colmean<<<128, 256, 0, stream>>>(aout, mbar);
  k_head<<<1, 256, 0, stream>>>(mbar, outW, outB, fcW, fcB, finW, finB, outp);
}
